// CellDecoder_12008728559660
// MI455X (gfx1250) — matrix operand order repaired and checked by reading, not compiled, not run on hardware
//
#include <hip/hip_runtime.h>
#include <hip/hip_bf16.h>

typedef __attribute__((ext_vector_type(16))) _Float16 v16h;
typedef __attribute__((ext_vector_type(8)))  _Float16 v8h;
typedef __attribute__((ext_vector_type(8)))  float    v8f;

#define CB 2
#define CT 512
#define CD 768
#define CE 256
#define CP 65536
#define CMLP 1536
#define CH1 768
#define CH2 384

// ---------------------------------------------------------------------------
// Transpose + convert: src f32 [K,N] row-major -> dst f16 [N,K] row-major.
// ---------------------------------------------------------------------------
__global__ void __launch_bounds__(256)
transpose_f32_to_f16(const float* __restrict__ src, _Float16* __restrict__ dst,
                     int K, int N)
{
    int i = blockIdx.x * blockDim.x + threadIdx.x;
    if (i >= K * N) return;
    int n = i / K;          // dst row (coalesced writes)
    int k = i - n * K;
    dst[(size_t)n * K + k] = (_Float16)src[(size_t)k * N + n];
}

// ---------------------------------------------------------------------------
// Span mean-pool + label-embedding concat -> ent_repr f16 [B*E, 2D]
// ---------------------------------------------------------------------------
__global__ void __launch_bounds__(256)
pool_concat_kernel(const float* __restrict__ hs,       // [B,T,D]
                   const float* __restrict__ emb,      // [5,D]
                   const int*   __restrict__ e_start,  // [B,E]
                   const int*   __restrict__ e_end,    // [B,E]
                   const int*   __restrict__ e_label,  // [B,E]
                   _Float16*    __restrict__ ent_repr) // [B*E, 2D]
{
    const int be  = blockIdx.x;        // b*E + e
    const int b   = be >> 8;           // E = 256
    const int s   = e_start[be];
    const int en  = e_end[be];
    const int lbl = e_label[be];
    int cnt = en - s; if (cnt < 1) cnt = 1;
    const float inv = 1.0f / (float)cnt;
    _Float16* dst = ent_repr + (size_t)be * CMLP;
    for (int c = threadIdx.x; c < CD; c += blockDim.x) {
        float acc = 0.0f;
        for (int t = s; t < en; ++t)
            acc += hs[((size_t)b * CT + t) * CD + c];
        dst[c]      = (_Float16)(acc * inv);
        dst[CD + c] = (_Float16)emb[(size_t)lbl * CD + c];
    }
}

// ---------------------------------------------------------------------------
// Generic WMMA GEMM: C[M,N] = A[M,K] (f16, row-major) x B (given as BT f16
// [N,K] row-major). Optional f32 bias + ReLU epilogue; writes f16 and/or f32.
// One wave32 computes one 16x16 tile with v_wmma_f32_16x16x32_f16.
// ---------------------------------------------------------------------------
__global__ void __launch_bounds__(128)
gemm_wmma_kernel(const _Float16* __restrict__ A,
                 const _Float16* __restrict__ BT,
                 const float*    __restrict__ bias,
                 _Float16*       __restrict__ O16,
                 float*          __restrict__ O32,
                 int M, int N, int K, int do_relu)
{
    const int lane     = threadIdx.x & 31;
    const int wave     = blockIdx.x * (blockDim.x >> 5) + (threadIdx.x >> 5);
    const int tiles_n  = N >> 4;
    const int tiles    = (M >> 4) * tiles_n;
    if (wave >= tiles) return;               // whole-wave uniform: EXEC all-1s inside
    const int tm = wave / tiles_n;
    const int tn = wave - tm * tiles_n;

    const int r16 = lane & 15;               // A: M within tile / B: N within tile
    const int kg  = lane >> 4;               // K sub-group select

    const _Float16* arow = A  + (size_t)(tm * 16 + r16) * K;
    const _Float16* brow = BT + (size_t)(tn * 16 + r16) * K;

    v8f c = {};
    for (int k0 = 0; k0 < K; k0 += 32) {
        // A layout (matches B): lanes 0-15 hold K = k0..k0+15,
        // lanes 16-31 hold K = k0+16..k0+31.
        union { v16h v; v8h h[2]; } a, bm;
        a.v    = *(const v16h*)(arow + k0 + kg * 16);
        // B (from [N,K] operand): lane n = lane&15, 16 contiguous K values,
        // lanes 0-15 -> K = k0..k0+15, lanes 16-31 -> K = k0+16..k0+31.
        bm.v   = *(const v16h*)(brow + k0 + kg * 16);
        __builtin_prefetch(arow + k0 + 64, 0, 1);
        __builtin_prefetch(brow + k0 + 64, 0, 1);
        c = __builtin_amdgcn_wmma_f32_16x16x32_f16(
                /*neg_a=*/false, a.v, /*neg_b=*/false, bm.v,
                /*c_mod=*/(short)0, c, /*reuse_a=*/false, /*reuse_b=*/false);
    }

    // C/D layout: VGPR r -> M = r + 8*(lane>>4), N = lane&15.
    const int ncol  = tn * 16 + r16;
    const int mbase = tm * 16 + kg * 8;
    const float bv = bias ? bias[ncol] : 0.0f;
    #pragma unroll
    for (int r = 0; r < 8; ++r) {
        float v = c[r] + bv;
        if (do_relu) v = v > 0.0f ? v : 0.0f;
        const size_t idx = (size_t)(mbase + r) * N + ncol;
        if (O16) O16[idx] = (_Float16)v;
        if (O32) O32[idx] = v;
    }
}

// ---------------------------------------------------------------------------
// Per-entity linear terms: linh[row,o] = head[row,:] . W_lin[0:384, o]
//                          lint[row,o] = tail[row,:] . W_lin[384:768, o]
// ---------------------------------------------------------------------------
__global__ void __launch_bounds__(256)
lin_terms_kernel(const _Float16* __restrict__ head,  // [B*E, H2] f16
                 const _Float16* __restrict__ tail,  // [B*E, H2] f16
                 const float*    __restrict__ W_lin, // [2*H2, 2]
                 float* __restrict__ linh,           // [B*E, 2]
                 float* __restrict__ lint)           // [B*E, 2]
{
    int i = blockIdx.x * blockDim.x + threadIdx.x;
    if (i >= CB * CE * 2) return;
    const int row = i >> 1;
    const int o   = i & 1;
    float ah = 0.0f, at = 0.0f;
    for (int k = 0; k < CH2; ++k) {
        ah += (float)head[(size_t)row * CH2 + k] * W_lin[k * 2 + o];
        at += (float)tail[(size_t)row * CH2 + k] * W_lin[(CH2 + k) * 2 + o];
    }
    linh[i] = ah;
    lint[i] = at;
}

// ---------------------------------------------------------------------------
// Final gather: out[b,p,o] = S[b,o,eh,et] + linh[b,eh,o] + lint[b,et,o] + b_lin[o]
// ---------------------------------------------------------------------------
__global__ void __launch_bounds__(256)
gather_out_kernel(const float* __restrict__ S,        // [B,2,E,E]
                  const float* __restrict__ linh,     // [B*E,2]
                  const float* __restrict__ lint,     // [B*E,2]
                  const float* __restrict__ b_lin,    // [2]
                  const int*   __restrict__ head_idx, // [B,P]
                  const int*   __restrict__ tail_idx, // [B,P]
                  float*       __restrict__ out)      // [B,P,2]
{
    int i = blockIdx.x * blockDim.x + threadIdx.x;
    if (i >= CB * CP) return;
    const int b  = i >> 16;                 // P = 65536
    const int eh = head_idx[i];
    const int et = tail_idx[i];
    const float h0 = linh[(b * CE + eh) * 2 + 0];
    const float h1 = linh[(b * CE + eh) * 2 + 1];
    const float t0 = lint[(b * CE + et) * 2 + 0];
    const float t1 = lint[(b * CE + et) * 2 + 1];
    const float s0 = S[((size_t)(b * 2 + 0) * CE + eh) * CE + et];
    const float s1 = S[((size_t)(b * 2 + 1) * CE + eh) * CE + et];
    out[(size_t)i * 2 + 0] = s0 + h0 + t0 + b_lin[0];
    out[(size_t)i * 2 + 1] = s1 + h1 + t1 + b_lin[1];
}

// ---------------------------------------------------------------------------
extern "C" void kernel_launch(void* const* d_in, const int* in_sizes, int n_in,
                              void* d_out, int out_size, void* d_ws, size_t ws_size,
                              hipStream_t stream)
{
    const float* hs   = (const float*)d_in[0];
    const float* emb  = (const float*)d_in[1];
    const float* Wh1  = (const float*)d_in[2];
    const float* bh1  = (const float*)d_in[3];
    const float* Wh2  = (const float*)d_in[4];
    const float* bh2  = (const float*)d_in[5];
    const float* Wt1  = (const float*)d_in[6];
    const float* bt1  = (const float*)d_in[7];
    const float* Wt2  = (const float*)d_in[8];
    const float* bt2  = (const float*)d_in[9];
    const float* Wbil = (const float*)d_in[10];
    const float* Wlin = (const float*)d_in[11];
    const float* blin = (const float*)d_in[12];
    const int* est  = (const int*)d_in[13];
    const int* een  = (const int*)d_in[14];
    const int* elb  = (const int*)d_in[15];
    const int* hidx = (const int*)d_in[16];
    const int* tidx = (const int*)d_in[17];
    float* out = (float*)d_out;

    // Workspace bump allocator (256B aligned).
    char* ws = (char*)d_ws;
    size_t off = 0;
    auto alloc = [&](size_t bytes) -> void* {
        off = (off + 255) & ~(size_t)255;
        void* p = ws + off;
        off += bytes;
        return p;
    };

    _Float16* Wh1T  = (_Float16*)alloc((size_t)CH1 * CMLP * 2);       // [768,1536]
    _Float16* Wh2T  = (_Float16*)alloc((size_t)CH2 * CH1 * 2);        // [384,768]
    _Float16* Wt1T  = (_Float16*)alloc((size_t)CH1 * CMLP * 2);
    _Float16* Wt2T  = (_Float16*)alloc((size_t)CH2 * CH1 * 2);
    _Float16* WbilT = (_Float16*)alloc((size_t)2 * CH2 * CH2 * 2);    // [2,384,384] (j,i)
    _Float16* entR  = (_Float16*)alloc((size_t)CB * CE * CMLP * 2);   // [512,1536]
    _Float16* Y1h   = (_Float16*)alloc((size_t)CB * CE * CH1 * 2);    // [512,768]
    _Float16* Y1t   = (_Float16*)alloc((size_t)CB * CE * CH1 * 2);
    _Float16* headF = (_Float16*)alloc((size_t)CB * CE * CH2 * 2);    // [512,384]
    _Float16* tailF = (_Float16*)alloc((size_t)CB * CE * CH2 * 2);
    _Float16* U     = (_Float16*)alloc((size_t)CB * 2 * CE * CH2 * 2);// [4,256,384]
    float*    S     = (float*)alloc((size_t)CB * 2 * CE * CE * 4);    // [4,256,256]
    float*    linh  = (float*)alloc((size_t)CB * CE * 2 * 4);
    float*    lint  = (float*)alloc((size_t)CB * CE * 2 * 4);
    (void)ws_size; (void)in_sizes; (void)n_in; (void)out_size;

    auto tr = [&](const float* src, _Float16* dst, int K, int N) {
        int n = K * N;
        transpose_f32_to_f16<<<(n + 255) / 256, 256, 0, stream>>>(src, dst, K, N);
    };
    auto gemm = [&](const _Float16* A, const _Float16* BT, const float* bias,
                    _Float16* O16, float* O32, int M, int N, int K, int relu) {
        int tiles  = (M / 16) * (N / 16);
        int blocks = (tiles + 3) / 4;        // 4 waves per 128-thread block
        gemm_wmma_kernel<<<blocks, 128, 0, stream>>>(A, BT, bias, O16, O32,
                                                     M, N, K, relu);
    };

    // 1) weight transposes + f16 conversion
    tr(Wh1, Wh1T, CMLP, CH1);
    tr(Wh2, Wh2T, CH1, CH2);
    tr(Wt1, Wt1T, CMLP, CH1);
    tr(Wt2, Wt2T, CH1, CH2);
    tr(Wbil,                 WbilT,               CH2, CH2);   // o = 0
    tr(Wbil + CH2 * CH2,     WbilT + CH2 * CH2,   CH2, CH2);   // o = 1

    // 2) span pooling + label concat
    pool_concat_kernel<<<CB * CE, 256, 0, stream>>>(hs, emb, est, een, elb, entR);

    // 3) FFNN (head + tail) via WMMA
    gemm(entR, Wh1T, bh1, Y1h,   nullptr, CB * CE, CH1, CMLP, 1);
    gemm(Y1h,  Wh2T, bh2, headF, nullptr, CB * CE, CH2, CH1,  1);
    gemm(entR, Wt1T, bt1, Y1t,   nullptr, CB * CE, CH1, CMLP, 1);
    gemm(Y1t,  Wt2T, bt2, tailF, nullptr, CB * CE, CH2, CH1,  1);

    // 4) bilinear score table S[b,o] = (head_b @ W_bil[o]) @ tail_b^T
    for (int b = 0; b < CB; ++b) {
        for (int o = 0; o < 2; ++o) {
            _Float16* Ubo = U + (size_t)(b * 2 + o) * CE * CH2;
            float*    Sbo = S + (size_t)(b * 2 + o) * CE * CE;
            gemm(headF + (size_t)b * CE * CH2, WbilT + (size_t)o * CH2 * CH2,
                 nullptr, Ubo, nullptr, CE, CH2, CH2, 0);
            gemm(Ubo, tailF + (size_t)b * CE * CH2,
                 nullptr, nullptr, Sbo, CE, CE, CH2, 0);
        }
    }

    // 5) per-entity linear terms
    {
        int n = CB * CE * 2;
        lin_terms_kernel<<<(n + 255) / 256, 256, 0, stream>>>(headF, tailF, Wlin,
                                                              linh, lint);
    }

    // 6) gather to output
    {
        int n = CB * CP;
        gather_out_kernel<<<(n + 255) / 256, 256, 0, stream>>>(S, linh, lint, blin,
                                                               hidx, tidx, out);
    }
}
